// Q_Transition_68719477414
// MI455X (gfx1250) — 8-bit matrix operand order repaired and checked by reading, compiled, not run on hardware
//
#include <hip/hip_runtime.h>
#include <stdint.h>


typedef int v8i __attribute__((ext_vector_type(8)));

constexpr int Bn   = 32;
constexpr int Cin  = 256;
constexpr int Cout = 128;
constexpr int Hh   = 56;
constexpr int Ww   = 56;
constexpr int HW   = Hh * Ww;
constexpr int Np   = Bn * HW;
constexpr int NX   = Bn * Cin * HW;
constexpr int OUTN = Bn * Cout * (Hh / 2) * (Ww / 2);

#define QMAXF 127.0f
#define EPS_S 1e-8f

__device__ __forceinline__ float clip8(float q) {
  return fminf(fmaxf(q, -128.0f), 127.0f);
}

__device__ __forceinline__ float loadScale(const unsigned* slot) {
  return fmaxf(__uint_as_float(*slot) / QMAXF, EPS_S);
}

__device__ __forceinline__ float blockMax(float v, float* sm) {
  int t = threadIdx.x;
  sm[t] = v;
  __syncthreads();
  for (int s = blockDim.x >> 1; s > 0; s >>= 1) {
    if (t < s) sm[t] = fmaxf(sm[t], sm[t + s]);
    __syncthreads();
  }
  return sm[0];
}

__global__ void k_init(unsigned* slots) {
  if (threadIdx.x < 8) slots[threadIdx.x] = 0u;
}

__global__ void k_bnfold(const float* __restrict__ gamma,
                         const float* __restrict__ beta,
                         const float* __restrict__ mean,
                         const float* __restrict__ var,
                         float* __restrict__ wq_bn,
                         float* __restrict__ b_fold) {
  int c = threadIdx.x;
  if (c < Cin) {
    float wf = gamma[c] * rsqrtf(var[c] + 1e-5f);
    float s  = fmaxf(fabsf(wf) / QMAXF, EPS_S);
    float wq = clip8(rintf(wf / s)) * s;
    wq_bn[c]  = wq;
    b_fold[c] = beta[c] - mean[c] * wf;
  }
}

__global__ void k_convq(const float* __restrict__ conv_w,
                        float* __restrict__ conv_s,
                        int8_t* __restrict__ qw8) {
  __shared__ float sm[256];
  int o = blockIdx.x, c = threadIdx.x;
  float w = conv_w[o * Cin + c];
  float m = blockMax(fabsf(w), sm);
  float s = fmaxf(m / QMAXF, EPS_S);
  qw8[o * Cin + c] = (int8_t)(int)clip8(rintf(w / s));
  if (c == 0) conv_s[o] = s;
}

__global__ void k_absmax_x(const float4* __restrict__ x4, unsigned* slots) {
  __shared__ float sm[256];
  float lmax = 0.0f;
  int n4 = NX / 4;
  for (int i = blockIdx.x * blockDim.x + threadIdx.x; i < n4;
       i += gridDim.x * blockDim.x) {
    float4 v = x4[i];
    lmax = fmaxf(lmax, fmaxf(fmaxf(fabsf(v.x), fabsf(v.y)),
                             fmaxf(fabsf(v.z), fabsf(v.w))));
  }
  float m = blockMax(lmax, sm);
  if (threadIdx.x == 0) atomicMax(&slots[0], __float_as_uint(m));
}

__global__ void k_absmax_y(const float4* __restrict__ x4,
                           const float* __restrict__ wq,
                           const float* __restrict__ bf,
                           unsigned* slots) {
  __shared__ float sm[256];
  float s1 = loadScale(&slots[0]);
  float lmax = 0.0f;
  int n4 = NX / 4;
  for (int i = blockIdx.x * blockDim.x + threadIdx.x; i < n4;
       i += gridDim.x * blockDim.x) {
    int idx   = i * 4;
    int plane = idx / HW;
    int c     = plane & (Cin - 1);
    float4 v = x4[i];
    float w = wq[c], b = bf[c];
    float y0 = clip8(rintf(v.x / s1)) * s1 * w + b;
    float y1 = clip8(rintf(v.y / s1)) * s1 * w + b;
    float y2 = clip8(rintf(v.z / s1)) * s1 * w + b;
    float y3 = clip8(rintf(v.w / s1)) * s1 * w + b;
    lmax = fmaxf(lmax, fmaxf(fmaxf(fabsf(y0), fabsf(y1)),
                             fmaxf(fabsf(y2), fabsf(y3))));
  }
  float m = blockMax(lmax, sm);
  if (threadIdx.x == 0) atomicMax(&slots[1], __float_as_uint(m));
}

__global__ void k_quant_store(const float* __restrict__ x,
                              const float* __restrict__ wq,
                              const float* __restrict__ bf,
                              uint8_t* __restrict__ act8T,
                              const unsigned* __restrict__ slots) {
  __shared__ uint8_t tile[64 * 260];
  float s1 = loadScale(&slots[0]);
  float s2 = loadScale(&slots[1]);
  int b   = blockIdx.x / (HW / 64);
  int hw0 = (blockIdx.x % (HW / 64)) * 64;
  int tx = threadIdx.x & 63;
  int ty = threadIdx.x >> 6;
  for (int c = ty; c < Cin; c += 4) {
    float xv = x[(size_t)(b * Cin + c) * HW + hw0 + tx];
    float x1 = clip8(rintf(xv / s1)) * s1;
    float y  = x1 * wq[c] + bf[c];
    float q  = fmaxf(clip8(rintf(y / s2)), 0.0f);
    tile[tx * 260 + c] = (uint8_t)(int)q;
  }
  __syncthreads();
  uint32_t* out32 = (uint32_t*)act8T;
  #pragma unroll
  for (int k = 0; k < 16; ++k) {
    int d  = threadIdx.x + k * 256;
    int pr = d >> 6;
    int cd = d & 63;
    const uint8_t* src = &tile[pr * 260 + cd * 4];
    uint32_t v = (uint32_t)src[0] | ((uint32_t)src[1] << 8) |
                 ((uint32_t)src[2] << 16) | ((uint32_t)src[3] << 24);
    out32[(size_t)(b * HW + hw0 + pr) * (Cin / 4) + cd] = v;
  }
}

__global__ void k_gemm_wmma(const int8_t* __restrict__ qw8,
                            const uint8_t* __restrict__ act8T,
                            const float* __restrict__ conv_s,
                            float* __restrict__ outf,
                            unsigned* __restrict__ slots) {
  __shared__ float sm[256];
  int lane = threadIdx.x & 31;
  int wave = threadIdx.x >> 5;
  int m0 = wave * 16;
  int p0 = blockIdx.x * 64;
  int nn = lane & 15;
  int hi = lane >> 4;

  const int8_t* Ap = qw8 + (size_t)(m0 + nn) * Cin;
  const uint8_t* Bp0 = act8T + (size_t)(p0 + nn) * Cin;

  v8i acc[4];
  #pragma unroll
  for (int t = 0; t < 4; ++t) acc[t] = (v8i){0, 0, 0, 0, 0, 0, 0, 0};

  #pragma unroll
  for (int k0 = 0; k0 < Cin; k0 += 64) {
    int2 a0 = *(const int2*)(Ap + k0 + 16 * hi);
    int2 a1 = *(const int2*)(Ap + k0 + 16 * hi + 8);
    int2 a2 = *(const int2*)(Ap + k0 + 16 * hi + 32);
    int2 a3 = *(const int2*)(Ap + k0 + 16 * hi + 40);
    v8i a; a[0]=a0.x; a[1]=a0.y; a[2]=a1.x; a[3]=a1.y;
           a[4]=a2.x; a[5]=a2.y; a[6]=a3.x; a[7]=a3.y;
    #pragma unroll
    for (int t = 0; t < 4; ++t) {
      const uint8_t* Bp = Bp0 + (size_t)(16 * t) * Cin;
      int4 b0 = *(const int4*)(Bp + k0 + 16 * hi);
      int4 b1 = *(const int4*)(Bp + k0 + 16 * hi + 32);
      v8i b; b[0]=b0.x; b[1]=b0.y; b[2]=b0.z; b[3]=b0.w;
             b[4]=b1.x; b[5]=b1.y; b[6]=b1.z; b[7]=b1.w;
      acc[t] = __builtin_amdgcn_wmma_i32_16x16x64_iu8(
           true, a,  false, b, acc[t],
           false,  false);
    }
  }

  float s2 = loadScale(&slots[1]);
  float lmax = 0.0f;
  #pragma unroll
  for (int r = 0; r < 8; ++r) {
    int m = m0 + hi * 8 + r;
    float sc = s2 * conv_s[m];
    float* orow = outf + (size_t)m * Np + p0 + nn;
    #pragma unroll
    for (int t = 0; t < 4; ++t) {
      float v = sc * (float)acc[t][r];
      orow[16 * t] = v;
      lmax = fmaxf(lmax, fabsf(v));
    }
  }
  float bm = blockMax(lmax, sm);
  if (threadIdx.x == 0) atomicMax(&slots[2], __float_as_uint(bm));
}

__global__ void k_pool(const float* __restrict__ outf,
                       float* __restrict__ dout,
                       unsigned* __restrict__ slots) {
  __shared__ float sm[256];
  float s3 = loadScale(&slots[2]);
  float lmax = 0.0f;
  for (int i = blockIdx.x * blockDim.x + threadIdx.x; i < OUTN;
       i += gridDim.x * blockDim.x) {
    int pw = i % 28;
    int ph = (i / 28) % 28;
    int o  = (i / 784) % Cout;
    int b  = i / (784 * Cout);
    const float* src = outf + (size_t)o * Np + b * HW + (2 * ph) * Ww + 2 * pw;
    float q0 = clip8(rintf(src[0]      / s3));
    float q1 = clip8(rintf(src[1]      / s3));
    float q2 = clip8(rintf(src[Ww]     / s3));
    float q3 = clip8(rintf(src[Ww + 1] / s3));
    float v = (q0 + q1 + q2 + q3) * 0.25f * s3;
    dout[i] = v;
    lmax = fmaxf(lmax, fabsf(v));
  }
  float bm = blockMax(lmax, sm);
  if (threadIdx.x == 0) atomicMax(&slots[3], __float_as_uint(bm));
}

__global__ void k_final(float* dout, const unsigned* __restrict__ slots) {
  float s4 = loadScale(&slots[3]);
  int i = blockIdx.x * blockDim.x + threadIdx.x;
  if (i < OUTN) dout[i] = clip8(rintf(dout[i] / s4)) * s4;
  if (i == 0) dout[OUTN] = s4;
}

extern "C" void kernel_launch(void* const* d_in, const int* in_sizes, int n_in,
                              void* d_out, int out_size, void* d_ws, size_t ws_size,
                              hipStream_t stream) {
  (void)in_sizes; (void)n_in; (void)out_size; (void)ws_size;
  const float* x      = (const float*)d_in[0];
  const float* gamma  = (const float*)d_in[1];
  const float* beta   = (const float*)d_in[2];
  const float* mean   = (const float*)d_in[3];
  const float* var    = (const float*)d_in[4];
  const float* conv_w = (const float*)d_in[5];
  float* out = (float*)d_out;

  char* ws = (char*)d_ws;
  unsigned* slots  = (unsigned*)ws;
  float*    conv_s = (float*)(ws + 64);
  float*    wq_bn  = (float*)(ws + 576);
  float*    b_fold = (float*)(ws + 1600);
  int8_t*   qw8    = (int8_t*)(ws + 4096);
  uint8_t*  act8T  = (uint8_t*)(ws + 65536);
  float*    outf   = (float*)(ws + 65536 + (size_t)Np * Cin);

  k_init      <<<1, 32, 0, stream>>>(slots);
  k_bnfold    <<<1, 256, 0, stream>>>(gamma, beta, mean, var, wq_bn, b_fold);
  k_convq     <<<Cout, 256, 0, stream>>>(conv_w, conv_s, qw8);
  k_absmax_x  <<<2048, 256, 0, stream>>>((const float4*)x, slots);
  k_absmax_y  <<<2048, 256, 0, stream>>>((const float4*)x, wq_bn, b_fold, slots);
  k_quant_store<<<Bn * (HW / 64), 256, 0, stream>>>(x, wq_bn, b_fold, act8T, slots);
  k_gemm_wmma <<<Np / 64, 256, 0, stream>>>(qw8, act8T, conv_s, outf, slots);
  k_pool      <<<2048, 256, 0, stream>>>(outf, out, slots);
  k_final     <<<(OUTN + 255) / 256, 256, 0, stream>>>(out, slots);
}
